// BasicBlock_72894184948219
// MI455X (gfx1250) — 8-bit matrix operand order repaired and checked by reading, compiled, not run on hardware
//
#include <hip/hip_runtime.h>
#include <hip/hip_bf16.h>

typedef __attribute__((ext_vector_type(16))) int   v16i;
typedef __attribute__((ext_vector_type(8)))  float v8f;

#define BATCH 32
#define CCH   256
#define HH    56
#define WW    56
#define HWH   (HH*WW)
#define NPIX  (BATCH*HWH)
#define KTAP  9
#define KTOT  (CCH*KTAP)

__device__ __forceinline__ unsigned char sign8(float v) {
    return v > 0.f ? 0x38 : (v < 0.f ? 0xB8 : 0x00);
}

__global__ __launch_bounds__(256) void prep_weights(
    const float* __restrict__ w, float* __restrict__ alpha,
    unsigned char* __restrict__ bfrag)
{
    const int o = blockIdx.x, tid = threadIdx.x;
    __shared__ float red[256];
    float s = 0.f;
    for (int idx = tid; idx < KTOT; idx += 256) s += fabsf(w[(size_t)o * KTOT + idx]);
    red[tid] = s;
    __syncthreads();
    for (int off = 128; off > 0; off >>= 1) {
        if (tid < off) red[tid] += red[tid + off];
        __syncthreads();
    }
    if (tid == 0) alpha[o] = red[0] * (1.0f / (float)KTOT);

    const int ntile = o >> 4, n = o & 15;
    for (int idx = tid; idx < KTOT; idx += 256) {
        const int i = idx / KTAP;
        const int t = idx % KTAP;
        const int kb  = i >> 7;
        const int k   = i & 127;
        const int g   = k >> 5, kin = k & 31;
        const int hi  = kin >> 4, kr = kin & 15;
        const int j   = kr >> 2, byt = kr & 3;
        const int lane = hi * 16 + n;
        const int dw   = g * 4 + j;
        const size_t off = ((size_t)((t * 2 + kb) * 16 + ntile)) * 2048
                         + (size_t)lane * 64 + dw * 4 + byt;
        bfrag[off] = sign8(w[(size_t)o * KTOT + idx]);
    }
}

__global__ __launch_bounds__(256) void binarize(
    const float* __restrict__ t,
    const float* __restrict__ sh1, const float* __restrict__ sh2,
    unsigned char* __restrict__ a1, unsigned char* __restrict__ a2)
{
    const size_t idx = (size_t)blockIdx.x * 256 + threadIdx.x;
    const int c = (int)(idx & 255);
    const size_t phw = idx >> 8;
    const size_t b = phw / HWH, hw = phw % HWH;
    const float v = t[((size_t)b * CCH + c) * HWH + hw];

    const int r  = c & 127;
    const int sw = (c & 128) + (((r >> 4) & 1) << 6) + ((r >> 5) << 4) + (r & 15);
    const size_t oidx = (phw << 8) + sw;
    a1[oidx] = sign8(v + sh1[c]);
    a2[oidx] = sign8(v + sh2[c]);
}

__global__ __launch_bounds__(256) void bingemm(
    const unsigned char* __restrict__ a1g, const unsigned char* __restrict__ a2g,
    const unsigned char* __restrict__ bfrag, const float* __restrict__ alpha,
    const float* __restrict__ sc, const float* __restrict__ gam,
    const float* __restrict__ bet, const float* __restrict__ mu,
    const float* __restrict__ var, const float* __restrict__ resid,
    float* __restrict__ out)
{
    __shared__ __align__(16) unsigned char bsm[2][16384];

    const int tid = threadIdx.x, lane = tid & 31, wv = tid >> 5;
    const int mg = wv >> 2, ng = wv & 3;
    const int mblockbase = blockIdx.x * 64;
    const int nb = blockIdx.y;

    const int bimg   = mblockbase / HWH;
    const int hwbase = mblockbase % HWH;

    v8f acc[2][2][2];
    const v8f vzero = {};
#pragma unroll
    for (int a = 0; a < 2; ++a)
#pragma unroll
        for (int b2 = 0; b2 < 2; ++b2)
#pragma unroll
            for (int c2 = 0; c2 < 2; ++c2) acc[a][b2][c2] = vzero;

    int ph[2], pw[2];
#pragma unroll
    for (int mt = 0; mt < 2; ++mt) {
        const int hw = hwbase + mg * 32 + mt * 16 + (lane & 15);
        ph[mt] = hw / WW;
        pw[mt] = hw % WW;
    }
    const long long imgbase   = (long long)bimg * HWH;
    const int       laneHiOff = (lane >> 4) << 6;
    const unsigned  ldsFrag0  = (unsigned)(ng * 2) * 2048 + (unsigned)lane * 64;

    {
        const int4* src = (const int4*)(bfrag + ((size_t)(nb * 8)) * 2048);
        int4* dst = (int4*)bsm[0];
#pragma unroll
        for (int r = 0; r < 4; ++r) dst[tid + r * 256] = src[tid + r * 256];
    }
    __syncthreads();

    for (int ks = 0; ks < 18; ++ks) {
        const int t   = ks >> 1;
        const int kb  = ks & 1;
        const int cur = ks & 1;

        int4 st[4];
        if (ks < 17) {
            const int kn = ks + 1;
            const int4* src = (const int4*)(bfrag +
                ((size_t)(((kn >> 1) * 2 + (kn & 1)) * 16 + nb * 8)) * 2048);
#pragma unroll
            for (int r = 0; r < 4; ++r) st[r] = src[tid + r * 256];
        }

        v16i bfr[2];
#pragma unroll
        for (int nt = 0; nt < 2; ++nt)
            bfr[nt] = *(const v16i*)(&bsm[cur][ldsFrag0 + (unsigned)nt * 2048]);

        const int dh = t / 3 - 1, dw = t % 3 - 1;
#pragma unroll
        for (int mt = 0; mt < 2; ++mt) {
            const int ih = ph[mt] + dh, iw = pw[mt] + dw;
            const bool inb = ((unsigned)ih < (unsigned)HH) && ((unsigned)iw < (unsigned)WW);
            const long long base = (((imgbase + (long long)ih * WW + iw) << 8)
                                   + (kb << 7) + laneHiOff);
            v16i afr[2];
#pragma unroll
            for (int v = 0; v < 2; ++v) {
                const unsigned char* ag = v ? a2g : a1g;
                const int4 zz = {0, 0, 0, 0};
#pragma unroll
                for (int q = 0; q < 4; ++q) {
                    const int4 u = inb ? *(const int4*)(ag + base + (long long)q * 16) : zz;
                    afr[v][4 * q]     = u.x;
                    afr[v][4 * q + 1] = u.y;
                    afr[v][4 * q + 2] = u.z;
                    afr[v][4 * q + 3] = u.w;
                }
            }
#pragma unroll
            for (int nt = 0; nt < 2; ++nt)
#pragma unroll
                for (int v = 0; v < 2; ++v)
                    acc[mt][nt][v] = __builtin_amdgcn_wmma_f32_16x16x128_fp8_fp8(
                        afr[v], bfr[nt], (short)0, acc[mt][nt][v], false, false);
        }

        if (ks < 17) {
            int4* dst = (int4*)bsm[cur ^ 1];
#pragma unroll
            for (int r = 0; r < 4; ++r) dst[tid + r * 256] = st[r];
            __syncthreads();
        }
    }

    const int nlocal = lane & 15;
    const int mh = (lane >> 4) << 3;
#pragma unroll
    for (int mt = 0; mt < 2; ++mt) {
#pragma unroll
        for (int nt = 0; nt < 2; ++nt) {
            const int o = nb * 128 + ng * 32 + nt * 16 + nlocal;
            const float al = alpha[o], s2 = sc[o];
            const float inv = rsqrtf(var[o] + 1e-5f);
            const float ga = gam[o], be = bet[o], mm = mu[o];
            const int hw0 = hwbase + mg * 32 + mt * 16 + mh;
            const size_t rowbase = ((size_t)bimg * CCH + o) * HWH + hw0;
#pragma unroll
            for (int r = 0; r < 8; ++r) {
                const size_t addr = rowbase + r;
                float val = al * (acc[mt][nt][0][r] + s2 * acc[mt][nt][1][r]);
                val = (val - mm) * inv * ga + be + resid[addr];
                val = fminf(1.f, fmaxf(-1.f, val));
                out[addr] = val;
            }
        }
    }
}

#define BFRAG_OFF 0u
#define BFRAG_SZ  (KTAP * 2 * 16 * 2048)
#define ALPHA_OFF (BFRAG_OFF + BFRAG_SZ)
#define ALPHA_SZ  (CCH * 4)
#define A1_OFF    (ALPHA_OFF + ALPHA_SZ)
#define A_SZ      ((size_t)NPIX * CCH)
#define A2_OFF    (A1_OFF + A_SZ)
#define INTER_OFF (A2_OFF + A_SZ)

extern "C" void kernel_launch(void* const* d_in, const int* in_sizes, int n_in,
                              void* d_out, int out_size, void* d_ws, size_t ws_size,
                              hipStream_t stream) {
    const float* x    = (const float*)d_in[0];
    const float* sh11 = (const float*)d_in[1];
    const float* sh12 = (const float*)d_in[2];
    const float* w1   = (const float*)d_in[3];
    const float* sc1  = (const float*)d_in[4];
    const float* g1   = (const float*)d_in[5];
    const float* b1   = (const float*)d_in[6];
    const float* m1   = (const float*)d_in[7];
    const float* v1   = (const float*)d_in[8];
    const float* sh21 = (const float*)d_in[9];
    const float* sh22 = (const float*)d_in[10];
    const float* w2   = (const float*)d_in[11];
    const float* sc2  = (const float*)d_in[12];
    const float* g2   = (const float*)d_in[13];
    const float* b2   = (const float*)d_in[14];
    const float* m2   = (const float*)d_in[15];
    const float* v2   = (const float*)d_in[16];

    unsigned char* ws = (unsigned char*)d_ws;
    unsigned char* bfrag = ws + BFRAG_OFF;
    float*         alpha = (float*)(ws + ALPHA_OFF);
    unsigned char* A1    = ws + A1_OFF;
    unsigned char* A2    = ws + A2_OFF;
    float*         inter = (float*)(ws + INTER_OFF);
    float*         outp  = (float*)d_out;

    const dim3 gblock(256);
    const dim3 gemm_grid(NPIX / 64, 2);

    prep_weights<<<CCH, gblock, 0, stream>>>(w1, alpha, bfrag);
    binarize<<<(unsigned)(A_SZ / 256), gblock, 0, stream>>>(x, sh11, sh12, A1, A2);
    bingemm<<<gemm_grid, gblock, 0, stream>>>(A1, A2, bfrag, alpha,
                                              sc1, g1, b1, m1, v1, x, inter);
    prep_weights<<<CCH, gblock, 0, stream>>>(w2, alpha, bfrag);
    binarize<<<(unsigned)(A_SZ / 256), gblock, 0, stream>>>(inter, sh21, sh22, A1, A2);
    bingemm<<<gemm_grid, gblock, 0, stream>>>(A1, A2, bfrag, alpha,
                                              sc2, g2, b2, m2, v2, inter, outp);
}
